// SpatialAxialAttention_19585050870311
// MI455X (gfx1250) — matrix operand order repaired and checked by reading, not compiled, not run on hardware
//
#include <hip/hip_runtime.h>
#include <hip/hip_bf16.h>
#include <math.h>

// ---------- types ----------
typedef __attribute__((ext_vector_type(16))) __bf16 v16bf;
typedef __attribute__((ext_vector_type(2)))  __bf16 v2bf;
typedef __attribute__((ext_vector_type(8)))  float  v8f;
typedef __attribute__((ext_vector_type(4)))  unsigned int v4u;
typedef __attribute__((ext_vector_type(8)))  int v8i;
typedef __attribute__((ext_vector_type(4)))  int v4i;

__device__ inline unsigned short f2bf(float f) {
    __bf16 b = (__bf16)f;
    return __builtin_bit_cast(unsigned short, b);
}
__device__ inline v2bf u2bf2(unsigned int u) {
    return __builtin_bit_cast(v2bf, u);
}

#define WMMA_BF16(a, b, c) \
    __builtin_amdgcn_wmma_f32_16x16x32_bf16(false, (a), false, (b), (short)0, (c), false, false)

// Problem constants: B=2,T=8,H=32,W=32,D=512, heads=8, head_dim=64
#define NTOK   16384
#define DMODEL 512
#define QKVF   1536
#define HEADS  8
#define HDIM   64
#define SEQ    1024   /* H*W per (b,t) slice */
#define BH     128    /* (B*T)*HEADS */

// ---------------------------------------------------------------------
// TDM: issue a 2-D tensor_load_to_lds (bf16 elements). Tile == tensor
// (tiles here are always fully in-bounds), row stride in elements.
// All descriptor words are wave-uniform -> SGPRs. D# per ISA §8.
// 6-arg builtin form (clang-23 lane): extra int32x8 group before cpol.
// ---------------------------------------------------------------------
__device__ inline void tdm_load_2d(unsigned int lds_off, const void* gaddr,
                                   unsigned int tile_d0, unsigned int tile_d1,
                                   unsigned int stride0)
{
    unsigned long long ga = (unsigned long long)(uintptr_t)gaddr;
    v4u g0;
    g0[0] = 1u;                                            // count=1, user mode
    g0[1] = lds_off;                                       // LDS byte address
    g0[2] = (unsigned int)(ga & 0xFFFFFFFFu);              // global_addr[31:0]
    g0[3] = (unsigned int)((ga >> 32) & 0x01FFFFFFu)       // global_addr[56:32]
            | 0x80000000u;                                 // type=2 ("image")
    v8i g1;
    g1[0] = (int)(1u << 16);                               // data_size=1 -> 2 bytes
    g1[1] = (int)((tile_d0 & 0xFFFFu) << 16);              // tensor_dim0[15:0]
    g1[2] = (int)((tile_d0 >> 16) | ((tile_d1 & 0xFFFFu) << 16)); // d0 hi | d1 lo
    g1[3] = (int)((tile_d1 >> 16) | (tile_d0 << 16));      // d1 hi | tile_dim0
    g1[4] = (int)(tile_d1 & 0xFFFFu);                      // tile_dim1 | tile_dim2=0
    g1[5] = (int)stride0;                                  // tensor_dim0_stride lo
    g1[6] = 0;                                             // stride0 hi | dim1_stride lo
    g1[7] = 0;                                             // dim1_stride hi
    v4i g2 = {0, 0, 0, 0};                                 // 2-D: groups 2/3 unused
    v4i g3 = {0, 0, 0, 0};
    v8i g4 = {0, 0, 0, 0, 0, 0, 0, 0};                     // unused extra group
    __builtin_amdgcn_tensor_load_to_lds(g0, g1, g2, g3, g4, 0);
}
__device__ inline unsigned int lds_offset(const void* p) {
    // generic LDS address: low 32 bits are the in-allocation byte offset
    return (unsigned int)(uintptr_t)p;
}

// =====================================================================
// Kernel 1: qkv = x @ w_qkv^T (bf16 WMMA, f32 accum), fused axial RoPE.
// Register tiling: each wave computes 4 M-tiles x 2 N-tiles (64x32).
// Writes q,k bf16 [bh][hw][d]; v bf16 transposed [bh][d][hw].
// =====================================================================
__global__ __launch_bounds__(256) void qkv_rope_kernel(
    const float* __restrict__ x, const float* __restrict__ wqkv,
    unsigned short* __restrict__ q_ws, unsigned short* __restrict__ k_ws,
    unsigned short* __restrict__ v_ws)
{
    const int lane  = threadIdx.x & 31;
    const int wave  = threadIdx.x >> 5;
    const int mbase = blockIdx.x * 64;
    const int nbase = (blockIdx.y * 16 + wave * 2) * 16;
    const int l15   = lane & 15;
    const int lhalf = lane >> 4;
    const int akb   = lhalf * 8;
    const int bkb   = lane & 16;

    const float* arow[4];
#pragma unroll
    for (int mt = 0; mt < 4; ++mt)
        arow[mt] = x + (size_t)(mbase + mt * 16 + l15) * DMODEL;
    const float* brow[2];
#pragma unroll
    for (int nt = 0; nt < 2; ++nt)
        brow[nt] = wqkv + (size_t)(nbase + nt * 16 + l15) * DMODEL;

    v8f zero = {};
    v8f acc[4][2];
#pragma unroll
    for (int mt = 0; mt < 4; ++mt)
#pragma unroll
        for (int nt = 0; nt < 2; ++nt) acc[mt][nt] = zero;

    for (int kk = 0; kk < DMODEL; kk += 32) {
        v16bf a[4], b[2];
#pragma unroll
        for (int mt = 0; mt < 4; ++mt)
#pragma unroll
            for (int i = 0; i < 8; ++i) {
                int k = kk + (akb << 1) + (i << 1);
                float2 p = *(const float2*)(arow[mt] + k);
                a[mt][2 * i] = (__bf16)p.x; a[mt][2 * i + 1] = (__bf16)p.y;
            }
#pragma unroll
        for (int nt = 0; nt < 2; ++nt)
#pragma unroll
            for (int j = 0; j < 8; ++j) {
                int k = kk + bkb + (j << 1);
                float2 p = *(const float2*)(brow[nt] + k);
                b[nt][2 * j] = (__bf16)p.x; b[nt][2 * j + 1] = (__bf16)p.y;
            }
#pragma unroll
        for (int mt = 0; mt < 4; ++mt)
#pragma unroll
            for (int nt = 0; nt < 2; ++nt)
                acc[mt][nt] = WMMA_BF16(a[mt], b[nt], acc[mt][nt]);
    }

    const float PI = 3.14159265358979323846f;
#pragma unroll
    for (int nt = 0; nt < 2; ++nt) {
        int n0   = nbase + nt * 16;
        int sec  = n0 / DMODEL;                 // 0=q 1=k 2=v (uniform per wave)
        int fmod = n0 % DMODEL;
        int h    = fmod / HDIM;
        int d    = (fmod % HDIM) + l15;
#pragma unroll
        for (int mt = 0; mt < 4; ++mt) {
#pragma unroll
            for (int r = 0; r < 8; ++r) {
                float val   = acc[mt][nt][r];
                int   token = mbase + mt * 16 + r + 8 * lhalf;
                int   bt    = token >> 10;
                int   hw    = token & 1023;
                int   bh    = bt * HEADS + h;
                if (sec < 2) {
                    float part = __shfl_xor(val, 1, 32);    // RoPE pair partner
                    int y = hw >> 5, xx = hw & 31;
                    float pos, basei;
                    if (d < 32) { pos = -1.0f + (2.0f / 31.0f) * (float)y;
                                  basei = (1.0f + (127.0f / 15.0f) * (float)(d >> 1)) * PI; }
                    else        { pos = -1.0f + (2.0f / 31.0f) * (float)xx;
                                  basei = (1.0f + (127.0f / 15.0f) * (float)((d - 32) >> 1)) * PI; }
                    float fr = pos * basei, sn, cs;
                    __sincosf(fr, &sn, &cs);
                    float res = (d & 1) ? (val * cs + part * sn) : (val * cs - part * sn);
                    unsigned short* dst = (sec == 0) ? q_ws : k_ws;
                    dst[((size_t)bh * SEQ + hw) * HDIM + d] = f2bf(res);
                } else {
                    v_ws[((size_t)bh * HDIM + d) * SEQ + hw] = f2bf(val);
                }
            }
        }
    }
}

// =====================================================================
// Kernel 2: flash attention per (bt,h). Grid (128, 8), block 128 = 4 waves,
// each wave owns 32 query rows (2 M-tiles). K/V tiles (32x64 / 64x32 bf16)
// are DMA'd into LDS by the Tensor Data Mover (double buffered, TENSORcnt),
// shared by all 4 waves. 8 WMMA for S + 8 WMMA for P@V per 32-key step.
// =====================================================================
__global__ __launch_bounds__(128) void attn_kernel(
    const unsigned short* __restrict__ q_ws, const unsigned short* __restrict__ k_ws,
    const unsigned short* __restrict__ v_ws, unsigned short* __restrict__ ao_ws)
{
    __shared__ __align__(16) unsigned short Kbuf[2][32 * HDIM];  // [key][d]
    __shared__ __align__(16) unsigned short Vbuf[2][HDIM * 32];  // [d][key]
    __shared__ __align__(16) unsigned short Pbuf[4][2][16 * 32];

    const int lane  = threadIdx.x & 31;
    const int wave  = threadIdx.x >> 5;
    const int bh    = blockIdx.x;
    const int qb    = blockIdx.y;
    const int bt    = bh >> 3, h = bh & 7;
    const int l15   = lane & 15;
    const int lhalf = lane >> 4;
    const int qbase = qb * 128 + wave * 32;     // 2 M-tiles of 16

    const unsigned short* qp = q_ws + (size_t)bh * SEQ * HDIM;
    const unsigned short* kp = k_ws + (size_t)bh * SEQ * HDIM;
    const unsigned short* vp = v_ws + (size_t)bh * HDIM * SEQ;

    // Q A-fragments: [mt][chunk]
    v16bf aq[2][2];
    {
        const int kb = lhalf * 8;
#pragma unroll
        for (int mt = 0; mt < 2; ++mt) {
            const unsigned short* qrow = qp + (size_t)(qbase + mt * 16 + l15) * HDIM;
#pragma unroll
            for (int i = 0; i < 8; ++i) {
                int dk = (kb << 1) + (i << 1);
                v2bf p0 = u2bf2(*(const unsigned int*)(qrow + dk));
                v2bf p1 = u2bf2(*(const unsigned int*)(qrow + 32 + dk));
                aq[mt][0][2 * i] = p0[0]; aq[mt][0][2 * i + 1] = p0[1];
                aq[mt][1][2 * i] = p1[0]; aq[mt][1][2 * i + 1] = p1[1];
            }
        }
    }

    float rm[2][8], rs[2][8];
#pragma unroll
    for (int mt = 0; mt < 2; ++mt)
#pragma unroll
        for (int r = 0; r < 8; ++r) { rm[mt][r] = -1e30f; rs[mt][r] = 0.0f; }
    v8f zero = {};
    v8f o[2][4];
#pragma unroll
    for (int mt = 0; mt < 2; ++mt)
#pragma unroll
        for (int j = 0; j < 4; ++j) o[mt][j] = zero;

    // Prologue: TDM-load first K/V tiles into buffer 0
    if (wave == 0) {
        tdm_load_2d(lds_offset(&Kbuf[0][0]), kp, HDIM, 32, HDIM);    // 32 keys x 64d
        tdm_load_2d(lds_offset(&Vbuf[0][0]), vp, 32, HDIM, SEQ);     // 64d x 32 keys
        __builtin_amdgcn_s_wait_tensorcnt(0);
    }
    __syncthreads();

    for (int kt = 0; kt < SEQ; kt += 32) {
        const int ib = (kt >> 5) & 1;
        const int haveNext = (kt + 32) < SEQ;
        // Kick off next tile's DMA while computing on the current one
        if (wave == 0 && haveNext) {
            tdm_load_2d(lds_offset(&Kbuf[ib ^ 1][0]), kp + (size_t)(kt + 32) * HDIM,
                        HDIM, 32, HDIM);
            tdm_load_2d(lds_offset(&Vbuf[ib ^ 1][0]), vp + (kt + 32),
                        32, HDIM, SEQ);
        }

        // ---- S = q @ k^T : 2 key tiles from LDS, shared across M-tiles ----
        v8f st[2][2];
#pragma unroll
        for (int t2 = 0; t2 < 2; ++t2) {
            const unsigned short* krow = &Kbuf[ib][(t2 * 16 + l15) * HDIM];
            const int dkb = lhalf * 16;
            v16bf b0, b1;
#pragma unroll
            for (int j = 0; j < 8; ++j) {
                v2bf p0 = u2bf2(*(const unsigned int*)(krow + dkb + 2 * j));
                v2bf p1 = u2bf2(*(const unsigned int*)(krow + 32 + dkb + 2 * j));
                b0[2 * j] = p0[0]; b0[2 * j + 1] = p0[1];
                b1[2 * j] = p1[0]; b1[2 * j + 1] = p1[1];
            }
#pragma unroll
            for (int mt = 0; mt < 2; ++mt) {
                v8f s = zero;
                s = WMMA_BF16(aq[mt][0], b0, s);
                s = WMMA_BF16(aq[mt][1], b1, s);
                st[mt][t2] = s;
            }
        }

        // ---- online softmax + stage P (bf16) into LDS ----
#pragma unroll
        for (int mt = 0; mt < 2; ++mt) {
            unsigned short* P = Pbuf[wave][mt];
#pragma unroll
            for (int r = 0; r < 8; ++r) {
                float s0 = st[mt][0][r] * 0.125f;     // 1/sqrt(64)
                float s1 = st[mt][1][r] * 0.125f;
                float mx = fmaxf(s0, s1);
#pragma unroll
                for (int msk = 1; msk < 16; msk <<= 1)
                    mx = fmaxf(mx, __shfl_xor(mx, msk, 32));
                float nm = fmaxf(rm[mt][r], mx);
                float sc = __expf(rm[mt][r] - nm);
                rm[mt][r] = nm;
                float p0 = __expf(s0 - nm), p1 = __expf(s1 - nm);
                float ts = p0 + p1;
#pragma unroll
                for (int msk = 1; msk < 16; msk <<= 1)
                    ts += __shfl_xor(ts, msk, 32);
                rs[mt][r] = rs[mt][r] * sc + ts;
#pragma unroll
                for (int j = 0; j < 4; ++j) o[mt][j][r] *= sc;
                int row = r + 8 * lhalf;
                P[row * 32 + l15]      = f2bf(p0);
                P[row * 32 + 16 + l15] = f2bf(p1);
            }
        }
        __syncthreads();

        // ---- P as A-fragments ----
        v16bf ap[2];
        {
            const int kb = lhalf * 8;
#pragma unroll
            for (int mt = 0; mt < 2; ++mt)
#pragma unroll
                for (int i = 0; i < 8; ++i) {
                    int kk2 = (kb << 1) + (i << 1);
                    v2bf p = u2bf2(*(const unsigned int*)(&Pbuf[wave][mt][l15 * 32 + kk2]));
                    ap[mt][2 * i] = p[0]; ap[mt][2 * i + 1] = p[1];
                }
        }

        // ---- O += P @ V : V fragments from LDS, reused across M-tiles ----
        const int kb3 = lane & 16;
#pragma unroll
        for (int j = 0; j < 4; ++j) {
            const unsigned short* vrow = &Vbuf[ib][(j * 16 + l15) * 32];
            v16bf bv;
#pragma unroll
            for (int jj = 0; jj < 8; ++jj) {
                v2bf p = u2bf2(*(const unsigned int*)(vrow + kb3 + 2 * jj));
                bv[2 * jj] = p[0]; bv[2 * jj + 1] = p[1];
            }
#pragma unroll
            for (int mt = 0; mt < 2; ++mt)
                o[mt][j] = WMMA_BF16(ap[mt], bv, o[mt][j]);
        }

        // All waves done reading buffer ib (and Pbuf) before it can be reused
        __syncthreads();
        if (wave == 0 && haveNext)
            __builtin_amdgcn_s_wait_tensorcnt(0);   // next buffer landed
        __syncthreads();                            // ... and is visible to all
    }

    // ---- normalize + store bf16 attention output [token][h*64+d] ----
#pragma unroll
    for (int mt = 0; mt < 2; ++mt)
#pragma unroll
        for (int r = 0; r < 8; ++r) {
            float inv = 1.0f / rs[mt][r];
            int   row = qbase + mt * 16 + r + 8 * lhalf;   // hw index
            size_t base = ((size_t)bt * SEQ + row) * DMODEL + h * HDIM;
            ao_ws[base +  0 + l15] = f2bf(o[mt][0][r] * inv);
            ao_ws[base + 16 + l15] = f2bf(o[mt][1][r] * inv);
            ao_ws[base + 32 + l15] = f2bf(o[mt][2][r] * inv);
            ao_ws[base + 48 + l15] = f2bf(o[mt][3][r] * inv);
        }
}

// =====================================================================
// Kernel 3: out = ao @ w_out^T + b_out (fp32 into d_out).
// Register tiling 4 M-tiles x 2 N-tiles per wave.
// =====================================================================
__global__ __launch_bounds__(256) void proj_kernel(
    const unsigned short* __restrict__ ao, const float* __restrict__ wout,
    const float* __restrict__ bout, float* __restrict__ out)
{
    const int lane  = threadIdx.x & 31;
    const int wave  = threadIdx.x >> 5;
    const int mbase = blockIdx.x * 64;
    const int nbase = (blockIdx.y * 16 + wave * 2) * 16;
    const int l15   = lane & 15;
    const int lhalf = lane >> 4;
    const int akb   = lhalf * 8;
    const int bkb   = lane & 16;

    const unsigned short* arow[4];
#pragma unroll
    for (int mt = 0; mt < 4; ++mt)
        arow[mt] = ao + (size_t)(mbase + mt * 16 + l15) * DMODEL;
    const float* brow[2];
#pragma unroll
    for (int nt = 0; nt < 2; ++nt)
        brow[nt] = wout + (size_t)(nbase + nt * 16 + l15) * DMODEL;

    v8f zero = {};
    v8f acc[4][2];
#pragma unroll
    for (int mt = 0; mt < 4; ++mt)
#pragma unroll
        for (int nt = 0; nt < 2; ++nt) acc[mt][nt] = zero;

    for (int kk = 0; kk < DMODEL; kk += 32) {
        v16bf a[4], b[2];
#pragma unroll
        for (int mt = 0; mt < 4; ++mt)
#pragma unroll
            for (int i = 0; i < 8; ++i) {
                int k = kk + (akb << 1) + (i << 1);
                v2bf p = u2bf2(*(const unsigned int*)(arow[mt] + k));
                a[mt][2 * i] = p[0]; a[mt][2 * i + 1] = p[1];
            }
#pragma unroll
        for (int nt = 0; nt < 2; ++nt)
#pragma unroll
            for (int j = 0; j < 8; ++j) {
                int k = kk + bkb + (j << 1);
                float2 p = *(const float2*)(brow[nt] + k);
                b[nt][2 * j] = (__bf16)p.x; b[nt][2 * j + 1] = (__bf16)p.y;
            }
#pragma unroll
        for (int mt = 0; mt < 4; ++mt)
#pragma unroll
            for (int nt = 0; nt < 2; ++nt)
                acc[mt][nt] = WMMA_BF16(a[mt], b[nt], acc[mt][nt]);
    }

#pragma unroll
    for (int nt = 0; nt < 2; ++nt) {
        int   dd   = nbase + nt * 16 + l15;
        float bias = bout[dd];
#pragma unroll
        for (int mt = 0; mt < 4; ++mt)
#pragma unroll
            for (int r = 0; r < 8; ++r) {
                int token = mbase + mt * 16 + r + 8 * lhalf;
                out[(size_t)token * DMODEL + dd] = acc[mt][nt][r] + bias;
            }
    }
}

// =====================================================================
extern "C" void kernel_launch(void* const* d_in, const int* in_sizes, int n_in,
                              void* d_out, int out_size, void* d_ws, size_t ws_size,
                              hipStream_t stream) {
    const float* x    = (const float*)d_in[0];   // (2,8,32,32,512)
    const float* wqkv = (const float*)d_in[1];   // (1536,512)
    const float* wout = (const float*)d_in[2];   // (512,512)
    const float* bout = (const float*)d_in[3];   // (512,)
    float* out = (float*)d_out;                  // (2,8,32,32,512)

    const size_t BUF = (size_t)NTOK * DMODEL;    // bf16 elements per buffer
    unsigned short* ws    = (unsigned short*)d_ws;
    unsigned short* q_ws  = ws;
    unsigned short* k_ws  = ws + BUF;
    unsigned short* v_ws  = ws + 2 * BUF;
    unsigned short* ao_ws = ws + 3 * BUF;

    qkv_rope_kernel<<<dim3(NTOK / 64, QKVF / 256), 256, 0, stream>>>(
        x, wqkv, q_ws, k_ws, v_ws);
    attn_kernel<<<dim3(BH, SEQ / 128), 128, 0, stream>>>(
        q_ws, k_ws, v_ws, ao_ws);
    proj_kernel<<<dim3(NTOK / 64, DMODEL / 256), 256, 0, stream>>>(
        ao_ws, wout, bout, out);
}
